// Split_LP_63745904607641
// MI455X (gfx1250) — matrix operand order repaired and checked by reading, not compiled, not run on hardware
//
#include <hip/hip_runtime.h>
#include <hip/hip_bf16.h>
#include <math.h>

// ---------------------------------------------------------------------------
// Split_LP on MI455X (gfx1250, wave32)
//   cluster_feats f: [B=128, K=512, D=256] f32 (L2-normalized rows)
//   query_feats   q: [B=128, D=256] f32
//   out P: [B, K, 2] f32  (solution of (I - alpha*S + eps) P = Y)
//
// Pipeline:
//   1) anchors_kernel : sim/argmin anchor selection; zero Dg
//   2) buildW_kernel  : W = exp(F F^T), diag=0 via f16 hi/lo split WMMA
//                       (v_wmma_f32_16x16x32_f16, 3 per k-chunk for ~fp32 acc)
//                       + column sums -> Dg
//   3) cg_kernel      : CG (fixed 160 iters; kappa<=199 -> err ~1e-10) on
//                       A x = x - alpha * d ∘ (W (d ∘ x)) + eps * sum(x)
// Workspace: W (128*512*512 f32 = 134.2MB) | Dg (128*512 f32) | anchors (128*2 i32)
// ---------------------------------------------------------------------------

typedef __attribute__((ext_vector_type(16))) _Float16 v16h;
typedef __attribute__((ext_vector_type(8)))  float    v8f;

#define K_CL   512
#define D_F    256
#define RB     128      // rows per WMMA block
#define LDH    264      // LDS row stride in f16 (padded vs 256 to spread banks)
#define ALPHA_LP 0.99f
#define EPS_LP 2.220446049250313e-16f
#define N_CG   160

// ---------------------------------------------------------------------------
// Kernel 1: anchor selection (+ zero Dg for this batch)
// ---------------------------------------------------------------------------
__global__ __launch_bounds__(256) void anchors_kernel(
    const float* __restrict__ f, const float* __restrict__ q,
    int* __restrict__ anchors, float* __restrict__ Dg)
{
  __shared__ float qs[D_F];
  __shared__ float sim[K_CL];
  __shared__ float rv[256];
  __shared__ int   ri[256];
  __shared__ int   s_a0;

  const int tid = threadIdx.x;
  const int b   = blockIdx.x;

  qs[tid] = q[(size_t)b * D_F + tid];
  Dg[(size_t)b * K_CL + tid]       = 0.0f;
  Dg[(size_t)b * K_CL + 256 + tid] = 0.0f;
  __syncthreads();

  const float4* f4 = (const float4*)f;

  // sim[k] = q . f_k
  for (int k = tid; k < K_CL; k += 256) {
    size_t base = ((size_t)b * K_CL + k) * (D_F / 4);
    float d = 0.0f;
    for (int j = 0; j < D_F / 4; ++j) {
      float4 v = f4[base + j];
      d += v.x * qs[4*j+0] + v.y * qs[4*j+1] + v.z * qs[4*j+2] + v.w * qs[4*j+3];
    }
    sim[k] = d;
  }
  __syncthreads();

  // argmin #1 (first-occurrence tie-break, like jnp.argmin)
  {
    float v0 = sim[tid], v1 = sim[tid + 256];
    float bv; int bi;
    if (v1 < v0) { bv = v1; bi = tid + 256; } else { bv = v0; bi = tid; }
    rv[tid] = bv; ri[tid] = bi; __syncthreads();
    for (int s = 128; s > 0; s >>= 1) {
      if (tid < s) {
        if (rv[tid+s] < rv[tid] || (rv[tid+s] == rv[tid] && ri[tid+s] < ri[tid])) {
          rv[tid] = rv[tid+s]; ri[tid] = ri[tid+s];
        }
      }
      __syncthreads();
    }
    if (tid == 0) s_a0 = ri[0];
    __syncthreads();
  }
  const int a0 = s_a0;
  __syncthreads();

  // sim = max(sim, f_{a0} . f_k); argmin #2
  qs[tid] = f[((size_t)b * K_CL + a0) * D_F + tid];
  __syncthreads();
  for (int k = tid; k < K_CL; k += 256) {
    size_t base = ((size_t)b * K_CL + k) * (D_F / 4);
    float d = 0.0f;
    for (int j = 0; j < D_F / 4; ++j) {
      float4 v = f4[base + j];
      d += v.x * qs[4*j+0] + v.y * qs[4*j+1] + v.z * qs[4*j+2] + v.w * qs[4*j+3];
    }
    sim[k] = fmaxf(sim[k], d);
  }
  __syncthreads();
  {
    float v0 = sim[tid], v1 = sim[tid + 256];
    float bv; int bi;
    if (v1 < v0) { bv = v1; bi = tid + 256; } else { bv = v0; bi = tid; }
    rv[tid] = bv; ri[tid] = bi; __syncthreads();
    for (int s = 128; s > 0; s >>= 1) {
      if (tid < s) {
        if (rv[tid+s] < rv[tid] || (rv[tid+s] == rv[tid] && ri[tid+s] < ri[tid])) {
          rv[tid] = rv[tid+s]; ri[tid] = ri[tid+s];
        }
      }
      __syncthreads();
    }
    if (tid == 0) { anchors[b*2 + 0] = a0; anchors[b*2 + 1] = ri[0]; }
  }
}

// ---------------------------------------------------------------------------
// Kernel 2: W = exp(F F^T) (diag 0) via hi/lo-split f16 WMMA; Dg column sums.
// Grid: (4 row-blocks, B batches). 256 threads = 8 waves.
// Dynamic LDS: Ahi|Alo|Bhi|Blo (RB*LDH f16 each) + colsum (K_CL f32)
// ---------------------------------------------------------------------------
__global__ __launch_bounds__(256) void buildW_kernel(
    const float* __restrict__ f, float* __restrict__ W, float* __restrict__ Dg)
{
  extern __shared__ char smem[];
  _Float16* Ahi = (_Float16*)smem;
  _Float16* Alo = Ahi + RB * LDH;
  _Float16* Bhi = Alo + RB * LDH;
  _Float16* Blo = Bhi + RB * LDH;
  float*  colsum = (float*)(Blo + RB * LDH);

  const int tid = threadIdx.x;
  const int b   = blockIdx.y;
  const int rb  = blockIdx.x;
  const int r0  = rb * RB;

  colsum[tid] = 0.0f;
  colsum[tid + 256] = 0.0f;

  const float4* f4 = (const float4*)f;

  // Load A block rows [r0, r0+RB) as hi/lo f16
  for (int i = tid; i < RB * (D_F / 4); i += 256) {
    int row = i >> 6, c4 = i & 63;
    float4 v = f4[((size_t)b * K_CL + r0 + row) * (D_F / 4) + c4];
    int o = row * LDH + c4 * 4;
    _Float16 h;
    h = (_Float16)v.x; Ahi[o+0] = h; Alo[o+0] = (_Float16)(v.x - (float)h);
    h = (_Float16)v.y; Ahi[o+1] = h; Alo[o+1] = (_Float16)(v.y - (float)h);
    h = (_Float16)v.z; Ahi[o+2] = h; Alo[o+2] = (_Float16)(v.z - (float)h);
    h = (_Float16)v.w; Ahi[o+3] = h; Alo[o+3] = (_Float16)(v.w - (float)h);
  }

  const int wv   = tid >> 5;
  const int lane = tid & 31;
  const int lrow = lane & 15;
  const int oh   = (lane >= 16) ? 8 : 0;    // A frag K-offset & C-matrix M-offset
  const int ob   = (lane >= 16) ? 16 : 0;   // unused; B frag uses oh (same k order as A)

  for (int cb = 0; cb < 4; ++cb) {
    const int c0 = cb * RB;
    __syncthreads();
    // Load B block rows [c0, c0+RB) as hi/lo f16
    for (int i = tid; i < RB * (D_F / 4); i += 256) {
      int row = i >> 6, c4 = i & 63;
      float4 v = f4[((size_t)b * K_CL + c0 + row) * (D_F / 4) + c4];
      int o = row * LDH + c4 * 4;
      _Float16 h;
      h = (_Float16)v.x; Bhi[o+0] = h; Blo[o+0] = (_Float16)(v.x - (float)h);
      h = (_Float16)v.y; Bhi[o+1] = h; Blo[o+1] = (_Float16)(v.y - (float)h);
      h = (_Float16)v.z; Bhi[o+2] = h; Blo[o+2] = (_Float16)(v.z - (float)h);
      h = (_Float16)v.w; Bhi[o+3] = h; Blo[o+3] = (_Float16)(v.w - (float)h);
    }
    __syncthreads();

    // 8x8 grid of 16x16 output tiles; wave-cyclic distribution
    for (int t = wv; t < 64; t += 8) {
      const int ti = t >> 3, tj = t & 7;
      const _Float16* pah = Ahi + (ti * 16 + lrow) * LDH;
      const _Float16* pal = Alo + (ti * 16 + lrow) * LDH;
      const _Float16* pbh = Bhi + (tj * 16 + lrow) * LDH;
      const _Float16* pbl = Blo + (tj * 16 + lrow) * LDH;

      v8f c = {};
#pragma unroll 4
      for (int kc = 0; kc < D_F; kc += 32) {
        v16h ah, al, bh, bl;
#pragma unroll
        for (int j = 0; j < 8; ++j) {
          ah[j]     = pah[kc + oh + j];
          ah[j + 8] = pah[kc + 16 + oh + j];
          al[j]     = pal[kc + oh + j];
          al[j + 8] = pal[kc + 16 + oh + j];
        }
#pragma unroll
        for (int j = 0; j < 8; ++j) {
          bh[j]     = pbh[kc + oh + j];
          bh[j + 8] = pbh[kc + 16 + oh + j];
          bl[j]     = pbl[kc + oh + j];
          bl[j + 8] = pbl[kc + 16 + oh + j];
        }
        // dot = hiA*hiB + hiA*loB + loA*hiB  (drop loA*loB ~ 2^-22)
        c = __builtin_amdgcn_wmma_f32_16x16x32_f16(false, ah, false, bh, (short)0, c, false, false);
        c = __builtin_amdgcn_wmma_f32_16x16x32_f16(false, ah, false, bl, (short)0, c, false, false);
        c = __builtin_amdgcn_wmma_f32_16x16x32_f16(false, al, false, bh, (short)0, c, false, false);
      }

      // C layout: vgpr i, lanes0-15 -> (M=i, N=lane); lanes16-31 -> (M=i+8, N=lane-16)
      const int colg = c0 + tj * 16 + lrow;
      const int rowb = r0 + ti * 16 + oh;
      float lsum = 0.0f;
#pragma unroll
      for (int i = 0; i < 8; ++i) {
        const int rowg = rowb + i;
        // branchless: unconditional fast exp (args in [-1,1]) + cndmask for diag
        const float e = __expf((float)c[i]);
        const float w = (rowg == colg) ? 0.0f : e;
        W[((size_t)b * K_CL + rowg) * K_CL + colg] = w;
        lsum += w;
      }
      atomicAdd(&colsum[colg], lsum);   // W symmetric: col sums == row sums
    }
  }
  __syncthreads();
  atomicAdd(&Dg[(size_t)b * K_CL + tid],       colsum[tid]);
  atomicAdd(&Dg[(size_t)b * K_CL + tid + 256], colsum[tid + 256]);
}

// ---------------------------------------------------------------------------
// Kernel 3: CG solve, 2 RHS simultaneously, fixed iteration count
// ---------------------------------------------------------------------------
__device__ __forceinline__ float sdiv(float n, float d) {
  return (fabsf(d) > 1e-38f) ? n / d : 0.0f;
}

__device__ __forceinline__ void bred2(float v0, float v1, float* red0, float* red1, int tid) {
  __syncthreads();                 // guard prior consumers of red*
  red0[tid] = v0; red1[tid] = v1; __syncthreads();
  for (int s = 128; s > 0; s >>= 1) {
    if (tid < s) { red0[tid] += red0[tid + s]; red1[tid] += red1[tid + s]; }
    __syncthreads();
  }
}

__global__ __launch_bounds__(256) void cg_kernel(
    const float* __restrict__ W, const float* __restrict__ Dg,
    const int* __restrict__ anchors, float* __restrict__ out)
{
  __shared__ float xs[K_CL][2], rr[K_CL][2], pp[K_CL][2], Ap[K_CL][2], tt[K_CL][2];
  __shared__ float dinv[K_CL];
  __shared__ float red0[256], red1[256];
  __shared__ float s_rs0, s_rs1, s_ac0, s_ac1, s_bt0, s_bt1;

  const int tid = threadIdx.x;
  const int b   = blockIdx.x;
  const int a0  = anchors[b*2 + 0];
  const int a1  = anchors[b*2 + 1];

  for (int k = tid; k < K_CL; k += 256) {
    dinv[k] = sqrtf(1.0f / (Dg[(size_t)b * K_CL + k] + EPS_LP));
    const float y0 = (k == a0) ? 1.0f : 0.0f;
    const float y1 = (k == a1) ? 1.0f : 0.0f;
    xs[k][0] = 0.0f; xs[k][1] = 0.0f;
    rr[k][0] = y0;   rr[k][1] = y1;
    pp[k][0] = y0;   pp[k][1] = y1;
  }
  {
    float n0 = 0.0f, n1 = 0.0f;
    for (int k = tid; k < K_CL; k += 256) { n0 += rr[k][0]*rr[k][0]; n1 += rr[k][1]*rr[k][1]; }
    bred2(n0, n1, red0, red1, tid);
    if (tid == 0) { s_rs0 = red0[0]; s_rs1 = red1[0]; }
  }

  const int wv = tid >> 5, lane = tid & 31;

  for (int it = 0; it < N_CG; ++it) {
    // t = dinv ∘ p ; sum(p) per RHS
    float sp0 = 0.0f, sp1 = 0.0f;
    for (int k = tid; k < K_CL; k += 256) {
      const float d = dinv[k];
      tt[k][0] = d * pp[k][0]; tt[k][1] = d * pp[k][1];
      sp0 += pp[k][0]; sp1 += pp[k][1];
    }
    bred2(sp0, sp1, red0, red1, tid);
    const float su0 = red0[0], su1 = red1[0];

    // Ap = p - alpha * dinv ∘ (W t) + eps * sum(p)
    for (int row = wv; row < K_CL; row += 8) {
      const float* wr = W + ((size_t)b * K_CL + row) * K_CL;
      __builtin_prefetch(W + ((size_t)b * K_CL + min(row + 8, K_CL - 1)) * K_CL + lane * 16, 0, 0);
      float u0 = 0.0f, u1 = 0.0f;
      for (int j = lane; j < K_CL; j += 32) {
        const float w = wr[j];
        u0 += w * tt[j][0]; u1 += w * tt[j][1];
      }
#pragma unroll
      for (int m = 16; m > 0; m >>= 1) {
        u0 += __shfl_xor(u0, m, 32);
        u1 += __shfl_xor(u1, m, 32);
      }
      if (lane == 0) {
        const float d = dinv[row];
        Ap[row][0] = pp[row][0] - ALPHA_LP * d * u0 + EPS_LP * su0;
        Ap[row][1] = pp[row][1] - ALPHA_LP * d * u1 + EPS_LP * su1;
      }
    }
    __syncthreads();

    // alpha = rs / (p . Ap)
    float q0 = 0.0f, q1 = 0.0f;
    for (int k = tid; k < K_CL; k += 256) {
      q0 += pp[k][0]*Ap[k][0]; q1 += pp[k][1]*Ap[k][1];
    }
    bred2(q0, q1, red0, red1, tid);
    if (tid == 0) { s_ac0 = sdiv(s_rs0, red0[0]); s_ac1 = sdiv(s_rs1, red1[0]); }
    __syncthreads();
    const float ac0 = s_ac0, ac1 = s_ac1;

    // x += alpha p ; r -= alpha Ap ; rs_new
    float n0 = 0.0f, n1 = 0.0f;
    for (int k = tid; k < K_CL; k += 256) {
      xs[k][0] += ac0 * pp[k][0]; xs[k][1] += ac1 * pp[k][1];
      rr[k][0] -= ac0 * Ap[k][0]; rr[k][1] -= ac1 * Ap[k][1];
      n0 += rr[k][0]*rr[k][0];    n1 += rr[k][1]*rr[k][1];
    }
    bred2(n0, n1, red0, red1, tid);
    if (tid == 0) {
      s_bt0 = sdiv(red0[0], s_rs0); s_bt1 = sdiv(red1[0], s_rs1);
      s_rs0 = red0[0];              s_rs1 = red1[0];
    }
    __syncthreads();
    const float bt0 = s_bt0, bt1 = s_bt1;

    for (int k = tid; k < K_CL; k += 256) {
      pp[k][0] = rr[k][0] + bt0 * pp[k][0];
      pp[k][1] = rr[k][1] + bt1 * pp[k][1];
    }
    __syncthreads();
  }

  for (int k = tid; k < K_CL; k += 256) {
    out[((size_t)b * K_CL + k) * 2 + 0] = xs[k][0];
    out[((size_t)b * K_CL + k) * 2 + 1] = xs[k][1];
  }
}

// ---------------------------------------------------------------------------
extern "C" void kernel_launch(void* const* d_in, const int* in_sizes, int n_in,
                              void* d_out, int out_size, void* d_ws, size_t ws_size,
                              hipStream_t stream) {
  (void)in_sizes; (void)n_in; (void)out_size; (void)ws_size;
  const int B = 128;
  const float* f = (const float*)d_in[0];   // [B, K, D]
  const float* q = (const float*)d_in[1];   // [B, D]
  float* out = (float*)d_out;               // [B, K, 2]

  float* W       = (float*)d_ws;                               // B*K*K
  float* Dg      = W + (size_t)B * K_CL * K_CL;                // B*K
  int*   anchors = (int*)(Dg + (size_t)B * K_CL);              // B*2

  anchors_kernel<<<B, 256, 0, stream>>>(f, q, anchors, Dg);

  const size_t shmem = (size_t)4 * RB * LDH * sizeof(_Float16) + (size_t)K_CL * sizeof(float);
  static bool attr_set = false;  // attribute set is not stream work; safe under capture
  if (!attr_set) {
    hipFuncSetAttribute((const void*)buildW_kernel,
                        hipFuncAttributeMaxDynamicSharedMemorySize, (int)shmem);
    attr_set = true;
  }
  buildW_kernel<<<dim3(4, B), 256, shmem, stream>>>(f, W, Dg);

  cg_kernel<<<B, 256, 0, stream>>>(W, Dg, anchors, out);
}
